// GRU_15719580303476
// MI455X (gfx1250) — matrix operand order repaired and checked by reading, not compiled, not run on hardware
//
#include <hip/hip_runtime.h>
#include <hip/hip_bf16.h>
#include <stdint.h>

typedef __attribute__((ext_vector_type(16))) _Float16 v16h;
typedef __attribute__((ext_vector_type(8)))  float    v8f;
typedef __attribute__((ext_vector_type(4)))  float    v4f;
typedef __attribute__((ext_vector_type(2)))  float    v2f;

#define HID  32
#define IND  4
#define TT   60
#define WARM 29

// gfx1250 hardware tanh (TRANS op). Prefer the clang builtin so the backend
// handles TRANS hazards/scheduling; fall back to inline asm with an explicit
// V_NOP (ISA rule: 1 independent op or V_NOP after TRANS before result use).
__device__ __forceinline__ float tanh_hw(float x) {
#if __has_builtin(__builtin_amdgcn_tanhf)
    return __builtin_amdgcn_tanhf(x);
#elif __has_builtin(__builtin_amdgcn_tanh_f32)
    return __builtin_amdgcn_tanh_f32(x);
#else
    float y;
    asm("v_tanh_f32 %0, %1\n\tv_nop" : "=v"(y) : "v"(x));
    return y;
#endif
}
// sigmoid(x) = 0.5 + 0.5*tanh(0.5x) -> single TRANS op instead of exp+rcp
__device__ __forceinline__ float sigm_hw(float x) {
    return fmaf(tanh_hw(0.5f * x), 0.5f, 0.5f);
}

union PKH { _Float16 f[2]; uint32_t u; };
union HBU { uint32_t u[8]; v16h v; };

__global__ __launch_bounds__(256) void gru_wmma_kernel(
    const float* __restrict__ input, const float* __restrict__ w_ih,
    const float* __restrict__ w_hh,  const float* __restrict__ b_ih,
    const float* __restrict__ b_hh,  const float* __restrict__ w_lin,
    const float* __restrict__ b_lin, float* __restrict__ out,
    float* __restrict__ h28, int B)
{
    const int  lane  = threadIdx.x & 31;
    const int  wid   = blockIdx.x * (blockDim.x >> 5) + (threadIdx.x >> 5);
    const int  mrow  = lane & 15;          // tile row (A) / batch column (B,C,D)
    const bool hi    = lane >= 16;
    const int  batch = wid * 16 + mrow;
    const int  bc    = (batch < B) ? batch : (B - 1);   // clamp for loads

    // ---------------- persistent A operands ----------------
    // f16 A 16x32 layout: lane<16: M=lane, K = {0..7,16..23}; lane>=16: K+=8
    v16h Ahh[6]; v2f Aih[6]; v16h Alin;
    #pragma unroll
    for (int t = 0; t < 6; ++t) {
        const int row = 16 * t + mrow;
        const float* wr = w_hh + row * HID;
        #pragma unroll
        for (int j = 0; j < 8; ++j) {
            const int k = (j < 4 ? 2 * j : 2 * j + 8) + (hi ? 8 : 0);
            Ahh[t][2 * j]     = (_Float16)wr[k];
            Ahh[t][2 * j + 1] = (_Float16)wr[k + 1];
        }
        // f32 A 16x4 layout: VGPR0 = K0 (lanes<16) / K2 (lanes>=16), VGPR1 = K1/K3
        Aih[t].x = w_ih[row * IND + (hi ? 2 : 0)];
        Aih[t].y = w_ih[row * IND + (hi ? 3 : 1)];
    }
    #pragma unroll
    for (int j = 0; j < 8; ++j) {   // w_lin padded to 16 rows (rows 4..15 = 0)
        const int k = (j < 4 ? 2 * j : 2 * j + 8) + (hi ? 8 : 0);
        Alin[2 * j]     = (_Float16)((mrow < 4) ? w_lin[mrow * HID + k]     : 0.f);
        Alin[2 * j + 1] = (_Float16)((mrow < 4) ? w_lin[mrow * HID + k + 1] : 0.f);
    }

    // ---------------- bias tiles in C/D layout ----------------
    // C/D: VGPR r holds M = r + 8*(lane>=16); value independent of lane%16
    v8f Crz[4], Cin[2], Chn[2];
    #pragma unroll
    for (int t = 0; t < 4; ++t)
        #pragma unroll
        for (int r = 0; r < 8; ++r) {
            const int row = 16 * t + (hi ? 8 : 0) + r;
            Crz[t][r] = b_ih[row] + b_hh[row];        // r,z gates: biases merge
        }
    #pragma unroll
    for (int t = 0; t < 2; ++t)
        #pragma unroll
        for (int r = 0; r < 8; ++r) {
            const int row = 16 * (4 + t) + (hi ? 8 : 0) + r;
            Cin[t][r] = b_ih[row];                    // n gate: keep separate
            Chn[t][r] = b_hh[row];                    // (b_hh sits inside r*h_n)
        }

    // hidden state, f32: h[8*tt + r] = hidden[16*tt + 8*hi + r]
    float h[16];
    #pragma unroll
    for (int i = 0; i < 16; ++i) h[i] = 0.f;
    float oPrev[4] = {0.f, 0.f, 0.f, 0.f};

    HBU hb;                                           // f16 B operand of h
    #pragma unroll
    for (int p = 0; p < 8; ++p) hb.u[p] = 0u;

    const float* inp  = input + (size_t)bc * (IND * TT) + (hi ? 2 : 0) * TT;
    float*       outp = out   + (size_t)bc * (IND * TT);
    const float bl0 = b_lin[0], bl1 = b_lin[1], bl2 = b_lin[2], bl3 = b_lin[3];

    for (int t = 0; t < TT; ++t) {
        // ---- x as f32 B 4x16: VGPR0 = x[c0], VGPR1 = x[c0+1], c0 = hi?2:0
        v2f xb;
        if (t < WARM) {
            // streamed once -> non-temporal so weights stay resident in L2
            xb.x = __builtin_nontemporal_load(inp + t);
            xb.y = __builtin_nontemporal_load(inp + t + TT);
        } else {
            const float s2 = __shfl_xor(oPrev[2], 16, 32);
            const float s3 = __shfl_xor(oPrev[3], 16, 32);
            xb.x = hi ? s2 : oPrev[0];
            xb.y = hi ? s3 : oPrev[1];
        }

        // gi = W_ih * x + bias   (exact f32, K=4 WMMA), chained into gh
        v8f g0 = __builtin_amdgcn_wmma_f32_16x16x4_f32(false, Aih[0], false, xb, (short)0, Crz[0], false, false);
        v8f g1 = __builtin_amdgcn_wmma_f32_16x16x4_f32(false, Aih[1], false, xb, (short)0, Crz[1], false, false);
        v8f g2 = __builtin_amdgcn_wmma_f32_16x16x4_f32(false, Aih[2], false, xb, (short)0, Crz[2], false, false);
        v8f g3 = __builtin_amdgcn_wmma_f32_16x16x4_f32(false, Aih[3], false, xb, (short)0, Crz[3], false, false);
        v8f gi4 = __builtin_amdgcn_wmma_f32_16x16x4_f32(false, Aih[4], false, xb, (short)0, Cin[0], false, false);
        v8f gi5 = __builtin_amdgcn_wmma_f32_16x16x4_f32(false, Aih[5], false, xb, (short)0, Cin[1], false, false);

        // gh (+gi for r,z) = W_hh * h  (f16 WMMA, f32 accumulate)
        v8f D0  = __builtin_amdgcn_wmma_f32_16x16x32_f16(false, Ahh[0], false, hb.v, (short)0, g0,     false, false);
        v8f D1  = __builtin_amdgcn_wmma_f32_16x16x32_f16(false, Ahh[1], false, hb.v, (short)0, g1,     false, false);
        v8f D2  = __builtin_amdgcn_wmma_f32_16x16x32_f16(false, Ahh[2], false, hb.v, (short)0, g2,     false, false);
        v8f D3  = __builtin_amdgcn_wmma_f32_16x16x32_f16(false, Ahh[3], false, hb.v, (short)0, g3,     false, false);
        v8f gh4 = __builtin_amdgcn_wmma_f32_16x16x32_f16(false, Ahh[4], false, hb.v, (short)0, Chn[0], false, false);
        v8f gh5 = __builtin_amdgcn_wmma_f32_16x16x32_f16(false, Ahh[5], false, hb.v, (short)0, Chn[1], false, false);

        // ---- gates + state update (lane-local; slot r <-> gate idx 16*tt+8*hi+r)
        #pragma unroll
        for (int r = 0; r < 8; ++r) {
            const float rg0 = sigm_hw(D0[r]);
            const float rg1 = sigm_hw(D1[r]);
            const float zg0 = sigm_hw(D2[r]);
            const float zg1 = sigm_hw(D3[r]);
            const float n0  = tanh_hw(fmaf(rg0, gh4[r], gi4[r]));
            const float n1  = tanh_hw(fmaf(rg1, gh5[r], gi5[r]));
            h[r]     = fmaf(zg0, h[r]     - n0, n0);   // (1-z)n + z*h
            h[8 + r] = fmaf(zg1, h[8 + r] - n1, n1);
        }

        // ---- rebuild f16 h B-operand (lane-local from own h; part[] unused)
        uint32_t own[8], part[8];
        #pragma unroll
        for (int p = 0; p < 8; ++p) {
            PKH pk; pk.f[0] = (_Float16)h[2 * p]; pk.f[1] = (_Float16)h[2 * p + 1];
            own[p] = pk.u;
        }
        #pragma unroll
        for (int p = 0; p < 8; ++p) part[p] = (uint32_t)__shfl_xor((int)own[p], 16, 32);
        #pragma unroll
        for (int j = 0; j < 4; ++j) hb.u[j] = own[j];
        #pragma unroll
        for (int j = 4; j < 8; ++j) hb.u[j] = own[j];

        // ---- out = W_lin * h_new + b_lin (from UPDATED h, per reference)
        v8f zc = {};
        v8f oD = __builtin_amdgcn_wmma_f32_16x16x32_f16(false, Alin, false, hb.v, (short)0, zc, false, false);
        oPrev[0] = oD[0] + bl0; oPrev[1] = oD[1] + bl1;
        oPrev[2] = oD[2] + bl2; oPrev[3] = oD[3] + bl3;

        if (!hi && batch < B) {   // lanes 0-15 hold out rows 0..3 for batch=lane
            __builtin_nontemporal_store(oPrev[0], outp + t);
            __builtin_nontemporal_store(oPrev[1], outp + t + TT);
            __builtin_nontemporal_store(oPrev[2], outp + t + 2 * TT);
            __builtin_nontemporal_store(oPrev[3], outp + t + 3 * TT);
        }
        if (t == WARM - 1 && batch < B) {
            float* ph = h28 + (size_t)batch * HID;
            const int off = hi ? 8 : 0;
            v4f a0 = { h[0],  h[1],  h[2],  h[3]  };
            v4f a1 = { h[4],  h[5],  h[6],  h[7]  };
            v4f a2 = { h[8],  h[9],  h[10], h[11] };
            v4f a3 = { h[12], h[13], h[14], h[15] };
            __builtin_nontemporal_store(a0, (v4f*)(ph + off));
            __builtin_nontemporal_store(a1, (v4f*)(ph + off + 4));
            __builtin_nontemporal_store(a2, (v4f*)(ph + 16 + off));
            __builtin_nontemporal_store(a3, (v4f*)(ph + 16 + off + 4));
        }
    }
}

extern "C" void kernel_launch(void* const* d_in, const int* in_sizes, int n_in,
                              void* d_out, int out_size, void* d_ws, size_t ws_size,
                              hipStream_t stream) {
    const float* input = (const float*)d_in[0];
    const float* w_ih  = (const float*)d_in[1];
    const float* w_hh  = (const float*)d_in[2];
    const float* b_ih  = (const float*)d_in[3];
    const float* b_hh  = (const float*)d_in[4];
    const float* w_lin = (const float*)d_in[5];
    const float* b_lin = (const float*)d_in[6];

    const int B = in_sizes[0] / (IND * TT);
    float* out = (float*)d_out;
    float* h28 = out + (size_t)B * IND * TT;

    const int batchesPerBlock = 8 * 16;   // 8 waves x 16 batch cols
    const int blocks = (B + batchesPerBlock - 1) / batchesPerBlock;
    gru_wmma_kernel<<<blocks, 256, 0, stream>>>(input, w_ih, w_hh, b_ih, b_hh,
                                                w_lin, b_lin, out, h28, B);
}
